// AdditiveDTMGP_77154792505994
// MI455X (gfx1250) — matrix operand order repaired and checked by reading, compiled, not run on hardware
//
#include <hip/hip_runtime.h>
#include <hip/hip_bf16.h>

typedef __attribute__((ext_vector_type(16))) _Float16 v16h;
typedef __attribute__((ext_vector_type(2)))  _Float16 h2;
typedef __attribute__((ext_vector_type(8)))  float    v8f;

// N=131072, D=8, M=63 (pad to 64 per feature -> K=512), widths 8,8,10 (pad to 16)
#define NSAMP   131072
#define NSTEP   16          // 512 / 32 (K per v_wmma_f32_16x16x32_f16)
#define BFRAG_ELEMS 8192    // NSTEP * 32 lanes * 16 halfs per layer

struct Params { const float* p[19]; };

__device__ __forceinline__ float softplusf(float x) { return log1pf(__expf(x)); }

// ---------------------------------------------------------------------------
// Prep: fold the (closed-form, upper-bidiagonal) Rinv into the reparameterized
// weights, emit them directly in the WMMA B-fragment layout (f16), compute
// bias sums and the analytic KL scalar.
// Rinv (OU kernel, uniform grid, rho = exp(-1/64)):
//   Rinv[m,m]   = (m==0) ? 1 : 1/sqrt(1-rho^2)
//   Rinv[m,m+1] = -rho / sqrt(1-rho^2)
// ---------------------------------------------------------------------------
__global__ __launch_bounds__(256) void dtmgp_prep(Params P,
                                                  _Float16* __restrict__ Bprep,
                                                  float* __restrict__ bsum,
                                                  float* __restrict__ klOut) {
    const int tid = threadIdx.x;
    const int wtab[3] = {8, 8, 10};
    const float rr   = __expf(-0.015625f);     // rho
    const float sC   = rsqrtf(1.0f - rr * rr); // 1/sqrt(1-rho^2)
    const float supC = -rr * sC;               // superdiagonal of Rinv

    // ---- folded weights in B-fragment layout ----
    // B layout (16-bit, 32x16): lane L holds col n=L&15; half=L>>4;
    // element e (0..15) of the per-lane v16h holds K = step*32 + (e<8 ? half*8+e : 16+half*8+(e-8)).
    for (int idx = tid; idx < 3 * BFRAG_ELEMS; idx += 256) {
        int l     = idx >> 13;
        int rem   = idx & (BFRAG_ELEMS - 1);
        int e     = rem & 15;
        int laneS = rem >> 4;          // step*32 + lane
        int lane  = laneS & 31;
        int s     = laneS >> 5;
        int hf    = lane >> 4;
        int n     = lane & 15;
        int K     = s * 32 + ((e < 8) ? (hf * 8 + e) : (16 + hf * 8 + (e - 8)));
        int d     = K >> 6, m = K & 63;
        int w     = wtab[l];
        float val = 0.0f;
        if (n < w && m < 63) {
            const float* Wmu  = P.p[1 + 6 * l];
            const float* Wrho = P.p[2 + 6 * l];
            const float* eW   = P.p[5 + 6 * l];
            int o0 = (d * 63 + m) * w + n;
            float ws0 = Wmu[o0] + softplusf(Wrho[o0]) * eW[o0];
            val = ((m == 0) ? 1.0f : sC) * ws0;
            if (m < 62) {
                int o1 = o0 + w;
                float ws1 = Wmu[o1] + softplusf(Wrho[o1]) * eW[o1];
                val += supC * ws1;
            }
        }
        Bprep[idx] = (_Float16)val;
    }

    // ---- bias sums (over features d), zero-padded to 16 cols ----
    if (tid < 48) {
        int l = tid >> 4, n = tid & 15;
        int w = wtab[l];
        float acc = 0.0f;
        if (n < w) {
            const float* bmu  = P.p[3 + 6 * l];
            const float* brho = P.p[4 + 6 * l];
            const float* eb   = P.p[6 + 6 * l];
            for (int d = 0; d < 8; ++d) {
                int o = d * w + n;
                acc += bmu[o] + softplusf(brho[o]) * eb[o];
            }
        }
        bsum[tid] = acc;
    }

    // ---- analytic KL over all variational params ----
    float acc = 0.0f;
    for (int l = 0; l < 3; ++l) {
        int w = wtab[l];
        const float* mu  = P.p[1 + 6 * l];
        const float* rho = P.p[2 + 6 * l];
        int nW = 504 * w;
        for (int i = tid; i < nW; i += 256) {
            float sg = softplusf(rho[i]);
            float m0 = mu[i];
            acc += 0.5f * (sg * sg + m0 * m0) - logf(sg) - 0.5f;
        }
        const float* bmu  = P.p[3 + 6 * l];
        const float* brho = P.p[4 + 6 * l];
        int nb = 8 * w;
        for (int i = tid; i < nb; i += 256) {
            float sg = softplusf(brho[i]);
            float m0 = bmu[i];
            acc += 0.5f * (sg * sg + m0 * m0) - logf(sg) - 0.5f;
        }
    }
    __shared__ float red[256];
    red[tid] = acc;
    __syncthreads();
    for (int off = 128; off > 0; off >>= 1) {
        if (tid < off) red[tid] += red[tid + off];
        __syncthreads();
    }
    if (tid == 0) klOut[0] = red[0];
}

// ---------------------------------------------------------------------------
// Main kernel. Each wave owns TWO 16-sample tiles (32 samples) so the two
// independent WMMA accumulation chains share every B fragment and the A-build
// of one tile fills the WMMA->VALU hazard window (4 co-exec slots) of the
// other -- no v_nops, half the B-fragment vmem traffic per sample.
//
// A fragment slot (j,p) sees K = c + 32*s  =>  d = s>>1, m = c + 32*(s&1):
// only two design points per slot, so 4 half2 constant tables per lane cover
// the whole kernel.  phi = exp(-|x-u|) = min(e^x * e^{-u}, e^{-x} * e^{u})
// -> 2x v_pk_mul_f16 + 1x v_pk_min_num_f16 per packed pair, 2 exps per
// (sample, feature, layer).  f16 inf/0 overflow is absorbed by min().
// ---------------------------------------------------------------------------
__global__ __launch_bounds__(256) void dtmgp_main(const float* __restrict__ x,
                                                  const _Float16* __restrict__ Bprep,
                                                  const float* __restrict__ bsum,
                                                  float* __restrict__ out) {
    __shared__ float hbuf[8][32][17];   // per-wave 2x16x16 tile, padded banks
    const int tid  = threadIdx.x;
    const int wid  = tid >> 5;
    const int lane = tid & 31;
    const int hf   = lane >> 4;
    const int r    = lane & 15;         // A-row / C-col index for this lane
    const int tileBase = (blockIdx.x * 8 + wid) * 32;

    // per-lane Laplace design-point constants in A-fragment order
    h2 cnLo[8], cnHi[8], cpLo[8], cpHi[8];
#pragma unroll
    for (int j = 0; j < 8; ++j) {
        int c = (j < 4) ? (hf * 8 + 2 * j) : (16 + hf * 8 + 2 * (j - 4));
#pragma unroll
        for (int p = 0; p < 2; ++p) {
            float uLo = (float)(c + p + 1) * 0.015625f;        // u_m, m = c+p
            float uHi = uLo + 0.5f;                            // u_{m+32}
            cnLo[j][p] = (_Float16)__expf(-uLo);
            cpLo[j][p] = (_Float16)__expf(uLo);
            cnHi[j][p] = (_Float16)__expf(-uHi);
            cpHi[j][p] = (_Float16)__expf(uHi);
        }
    }

    float xv0[8], xv1[8];
    {
        const float* x0 = x + (size_t)(tileBase + r) * 8;
        const float* x1 = x + (size_t)(tileBase + 16 + r) * 8;
#pragma unroll
        for (int d = 0; d < 8; ++d) { xv0[d] = x0[d]; xv1[d] = x1[d]; }
    }

#pragma unroll
    for (int l = 0; l < 3; ++l) {
        // 2 exps per feature per tile per layer
        h2 ep0[8], em0[8], ep1[8], em1[8];
#pragma unroll
        for (int d = 0; d < 8; ++d) {
            _Float16 a1 = (_Float16)__expf(xv0[d]);
            _Float16 a0 = (_Float16)__expf(-xv0[d]);
            _Float16 b1 = (_Float16)__expf(xv1[d]);
            _Float16 b0 = (_Float16)__expf(-xv1[d]);
            ep0[d] = (h2){a1, a1}; em0[d] = (h2){a0, a0};
            ep1[d] = (h2){b1, b1}; em1[d] = (h2){b0, b0};
        }

        const _Float16* Bl = Bprep + l * BFRAG_ELEMS;
        v8f c0 = {}, c1 = {};
#pragma unroll
        for (int s = 0; s < NSTEP; ++s) {
            const int d  = s >> 1;
            const int hi = s & 1;
            v16h b = *(const v16h*)(Bl + (s * 32 + lane) * 16);

            v16h a0;
#pragma unroll
            for (int j = 0; j < 8; ++j) {
                h2 t = __builtin_elementwise_min(
                           ep0[d] * (hi ? cnHi[j] : cnLo[j]),
                           em0[d] * (hi ? cpHi[j] : cpLo[j]));
                a0[2 * j] = t[0]; a0[2 * j + 1] = t[1];
            }
            c0 = __builtin_amdgcn_wmma_f32_16x16x32_f16(
                     false, a0, false, b, (short)0, c0, false, false);

            v16h a1;
#pragma unroll
            for (int j = 0; j < 8; ++j) {
                h2 t = __builtin_elementwise_min(
                           ep1[d] * (hi ? cnHi[j] : cnLo[j]),
                           em1[d] * (hi ? cpHi[j] : cpLo[j]));
                a1[2 * j] = t[0]; a1[2 * j + 1] = t[1];
            }
            c1 = __builtin_amdgcn_wmma_f32_16x16x32_f16(
                     false, a1, false, b, (short)0, c1, false, false);
        }

        float bb = bsum[l * 16 + r];    // bias for output column r (0 if padded)
#pragma unroll
        for (int v = 0; v < 8; ++v) { c0[v] += bb; c1[v] += bb; }

        if (l < 2) {
            // transpose C/D layout (row = v+8*hf, col = r) -> A layout (row = lane)
            __syncthreads();
#pragma unroll
            for (int v = 0; v < 8; ++v) {
                hbuf[wid][v + 8 * hf][r]      = c0[v];
                hbuf[wid][16 + v + 8 * hf][r] = c1[v];
            }
            __syncthreads();
#pragma unroll
            for (int d = 0; d < 8; ++d) {
                xv0[d] = hbuf[wid][r][d];
                xv1[d] = hbuf[wid][16 + r][d];
            }
        } else {
            if (r < 10) {
#pragma unroll
                for (int v = 0; v < 8; ++v) {
                    out[(size_t)(tileBase + v + 8 * hf) * 10 + r]      = c0[v];
                    out[(size_t)(tileBase + 16 + v + 8 * hf) * 10 + r] = c1[v];
                }
            }
        }
    }
}

extern "C" void kernel_launch(void* const* d_in, const int* in_sizes, int n_in,
                              void* d_out, int out_size, void* d_ws, size_t ws_size,
                              hipStream_t stream) {
    (void)in_sizes; (void)ws_size;
    Params P;
    for (int i = 0; i < 19 && i < n_in; ++i) P.p[i] = (const float*)d_in[i];

    _Float16* Bprep = (_Float16*)d_ws;                                    // 48 KB
    float*    bsum  = (float*)((char*)d_ws + 3 * BFRAG_ELEMS * sizeof(_Float16));
    float*    out   = (float*)d_out;
    float*    klOut = out + (out_size - 1);   // [N*10] outputs, then KL scalar

    dtmgp_prep<<<1, 256, 0, stream>>>(P, Bprep, bsum, klOut);
    dtmgp_main<<<NSAMP / 256, 256, 0, stream>>>((const float*)d_in[0], Bprep, bsum, out);
}
